// QModel_Bounded_88691074663169
// MI455X (gfx1250) — matrix operand order repaired and checked by reading, not compiled, not run on hardware
//
#include <hip/hip_runtime.h>

typedef _Float16 v16h __attribute__((ext_vector_type(16)));
typedef _Float16 h8   __attribute__((ext_vector_type(8)));
typedef float    v8f  __attribute__((ext_vector_type(8)));
typedef float    f4   __attribute__((ext_vector_type(4)));
typedef unsigned int u4 __attribute__((ext_vector_type(4)));

union U8 { h8 v; _Float16 h[8]; unsigned int u[4]; };

// ---- fragment-packed weight layout (built by k_pack_frags) ----
// 16 k-blocks; per k-block 42 fragments of 1KB (32 lanes x 16 halves):
//   local  0..23 : layer-1 A-frags   [mlp m(0..2)][tile tl(0..1)][k4(0..3)]
//   local 24     : nu_w2 A-frag
//   local 25     : v_w2  A-frag (rows 1..15 zero-padded)
//   local 26..41 : p_w2  A-frags [t2(0..15)]
#define FRAGS_PER_KB 42
#define HALVES_PER_KB (FRAGS_PER_KB * 512)   // 21504 halves = 43008 B
#define CHUNKS_PER_KB (FRAGS_PER_KB * 64)    // 2688 x 16B

__device__ __forceinline__ v8f wmma_f16(v16h a, v16h b, v8f c) {
  return __builtin_amdgcn_wmma_f32_16x16x32_f16(false, a, false, b, (short)0, c,
                                                false, false);
}

// base already includes lane*16 halves; offset is a compile-time immediate
__device__ __forceinline__ v16h lfrag2(const _Float16* base, int local) {
  return *(const v16h*)(base + local * 512);
}

// stage one k-block's fragment slab (43008 B) global -> LDS
__device__ __forceinline__ void stage(const _Float16* __restrict__ gsrc,
                                      _Float16* sdst, int tid) {
  const u4* s = (const u4*)gsrc;
  u4* d = (u4*)sdst;
  for (int cc = tid; cc < CHUNKS_PER_KB; cc += 256) d[cc] = s[cc];
}

// Layer-1 for one MLP: grouped 8-fragment LDS load, 8 WMMAs, bias+ReLU,
// then assemble the layer-2 B-fragment via one cross-half shuffle.
__device__ __forceinline__ v16h mlp_l1(const _Float16* base, int m,
                                       const float* __restrict__ b1,
                                       int c, const v16h* sfrag, int hf) {
  v16h f[8];
#pragma unroll
  for (int i = 0; i < 8; ++i) f[i] = lfrag2(base, m * 8 + i);
  v8f c0, c1;
#pragma unroll
  for (int g = 0; g < 8; ++g) { c0[g] = 0.f; c1[g] = 0.f; }
#pragma unroll
  for (int k4 = 0; k4 < 4; ++k4) {
    c0 = wmma_f16(f[k4],     sfrag[k4], c0);
    c1 = wmma_f16(f[4 + k4], sfrag[k4], c1);
  }
  U8 t0, t1, x0, x1;
#pragma unroll
  for (int g = 0; g < 8; ++g) {
    float h0 = fmaxf(c0[g] + b1[c + hf * 8 + g], 0.f);
    float h1 = fmaxf(c1[g] + b1[c + 16 + hf * 8 + g], 0.f);
    t0.h[g] = (_Float16)h0;
    t1.h[g] = (_Float16)h1;
  }
#pragma unroll
  for (int w = 0; w < 4; ++w) {
    x0.u[w] = (unsigned)__shfl_xor((int)t0.u[w], 16, 32);
    x1.u[w] = (unsigned)__shfl_xor((int)t1.u[w], 16, 32);
  }
  U8 lo, hi;
#pragma unroll
  for (int w = 0; w < 4; ++w) {
    lo.u[w] = hf ? x1.u[w] : t0.u[w];
    hi.u[w] = hf ? t1.u[w] : x0.u[w];
  }
  return __builtin_shufflevector(lo.v, hi.v, 0, 1, 2, 3, 4, 5, 6, 7,
                                             8, 9, 10, 11, 12, 13, 14, 15);
}

// One k-block: optional stage of the next slab, 3 layer-1 MLPs, heads.
__device__ __forceinline__ void kb_body(
    const _Float16* curbase,                 // smem(+43008B) + lane*32B
    _Float16* stage_dst,                     // other buffer base, or nullptr
    const _Float16* __restrict__ stage_src,
    const _Float16* __restrict__ pref_src,
    int c, int tid, int hf,
    const v16h* sfrag,
    const float* __restrict__ nu_b1, const float* __restrict__ v_b1,
    const float* __restrict__ p_b1,
    v8f& nu_acc, v8f& v_acc, v8f* pacc) {
  if (stage_dst) {
    if (pref_src)   // warm L2 one slab ahead of the staged one
      __builtin_prefetch(pref_src + tid * 64, 0, 1);
    stage(stage_src, stage_dst, tid);
  }

  v16h bnu = mlp_l1(curbase, 0, nu_b1, c, sfrag, hf);
  v16h bvv = mlp_l1(curbase, 1, v_b1,  c, sfrag, hf);
  v16h bpp = mlp_l1(curbase, 2, p_b1,  c, sfrag, hf);

  v16h f24 = lfrag2(curbase, 24);
  v16h f25 = lfrag2(curbase, 25);
  nu_acc = wmma_f16(f24, bnu, nu_acc);
  v_acc  = wmma_f16(f25, bvv, v_acc);

#pragma unroll
  for (int grp = 0; grp < 2; ++grp) {
    v16h fp[8];
#pragma unroll
    for (int i = 0; i < 8; ++i) fp[i] = lfrag2(curbase, 26 + grp * 8 + i);
#pragma unroll
    for (int i = 0; i < 8; ++i)
      pacc[grp * 8 + i] = wmma_f16(fp[i], bpp, pacc[grp * 8 + i]);
  }
  __syncthreads();
}

__global__ __launch_bounds__(256) void qmodel_fused(
    const float* __restrict__ state, const float* __restrict__ action,
    const _Float16* __restrict__ wpack,
    const float* __restrict__ nu_b1, const float* __restrict__ nu_b2,
    const float* __restrict__ v_b1,  const float* __restrict__ v_b2,
    const float* __restrict__ p_b1,  const float* __restrict__ p_b2,
    float* __restrict__ out) {
  const int tid  = threadIdx.x;
  const int lane = tid & 31;
  const int wave = tid >> 5;                       // 0..7
  const int rb   = (blockIdx.x * 8 + wave) * 16;   // 16 batch rows per wave
  const int l    = lane & 15;
  const int hf   = lane >> 4;

  extern __shared__ _Float16 smem[];               // 2 x 43008 B double buffer
  _Float16* buf0 = smem;
  _Float16* buf1 = smem + HALVES_PER_KB;
  const _Float16* base0 = buf0 + lane * 16;        // per-lane bases: all frag
  const _Float16* base1 = buf1 + lane * 16;        // loads use imm offsets

  // state^T B-fragments for the 4 k-blocks of K=128 (f32 -> f16 in regs)
  v16h sfrag[4];
  const float* srow = state + (size_t)(rb + l) * 128 + hf * 16;
#pragma unroll
  for (int k4 = 0; k4 < 4; ++k4) {
    const f4* p = (const f4*)(srow + 32 * k4);
    f4 a = p[0], b = p[1], cc = p[2], d = p[3];
    v16h t;
#pragma unroll
    for (int i = 0; i < 4; ++i) {
      t[i]      = (_Float16)a[i];
      t[4 + i]  = (_Float16)b[i];
      t[8 + i]  = (_Float16)cc[i];
      t[12 + i] = (_Float16)d[i];
    }
    sfrag[k4] = t;
  }

  // layer-2 accumulators (out^T, C-layout)
  v8f nu_acc, v_acc, pacc[16];
#pragma unroll
  for (int g = 0; g < 8; ++g) { nu_acc[g] = 0.f; v_acc[g] = 0.f; }
#pragma unroll
  for (int t2 = 0; t2 < 16; ++t2)
#pragma unroll
    for (int g = 0; g < 8; ++g) pacc[t2][g] = 0.f;

  stage(wpack, buf0, tid);
  __syncthreads();

  // k-loop unrolled by 2 so buffer bases are compile-time per half
  for (int kb2 = 0; kb2 < 7; ++kb2) {
    const int kb = 2 * kb2;
    kb_body(base0, buf1, wpack + (size_t)(kb + 1) * HALVES_PER_KB,
            wpack + (size_t)(kb + 2) * HALVES_PER_KB,
            kb * 32, tid, hf, sfrag, nu_b1, v_b1, p_b1, nu_acc, v_acc, pacc);
    kb_body(base1, buf0, wpack + (size_t)(kb + 2) * HALVES_PER_KB,
            wpack + (size_t)(kb + 3) * HALVES_PER_KB,
            (kb + 1) * 32, tid, hf, sfrag, nu_b1, v_b1, p_b1, nu_acc, v_acc, pacc);
  }
  kb_body(base0, buf1, wpack + (size_t)15 * HALVES_PER_KB, nullptr,
          14 * 32, tid, hf, sfrag, nu_b1, v_b1, p_b1, nu_acc, v_acc, pacc);
  kb_body(base1, nullptr, nullptr, nullptr,
          15 * 32, tid, hf, sfrag, nu_b1, v_b1, p_b1, nu_acc, v_acc, pacc);

  // ---- epilogue (per lane: batch row n = rb + l) ----
  float no[8], po[8];
#pragma unroll
  for (int g = 0; g < 8; ++g) {
    no[g] = nu_acc[g];                     // comps g + 8*hf
    po[g] = __shfl_xor(no[g], 16, 32);     // comps g + 8*(1-hf)
  }
  float nuf[16];
#pragma unroll
  for (int g = 0; g < 8; ++g) {
    nuf[g]     = hf ? po[g] : no[g];
    nuf[g + 8] = hf ? no[g] : po[g];
  }
#pragma unroll
  for (int i = 0; i < 16; ++i) nuf[i] += nu_b2[i];

  float act[16];
  const f4* arow = (const f4*)(action + (size_t)(rb + l) * 16);
#pragma unroll
  for (int w = 0; w < 4; ++w) {
    f4 a = arow[w];
#pragma unroll
    for (int i = 0; i < 4; ++i) act[4 * w + i] = a[i];
  }
  float anu[16], mnu[16];
#pragma unroll
  for (int i = 0; i < 16; ++i) {
    anu[i] = act[i] - nuf[i];
    float mu = fminf(fmaxf(nuf[i], -1.f), 1.f);   // interval transform == clip
    mnu[i] = mu - nuf[i];
  }

  // qa = ||L^T a||^2 : lane owns columns j = g + 8*hf; P-tile t is row t of L
  float ca[8], cm[8];
#pragma unroll
  for (int g = 0; g < 8; ++g) { ca[g] = 0.f; cm[g] = 0.f; }
#pragma unroll
  for (int t = 0; t < 16; ++t) {
#pragma unroll
    for (int g = 0; g < 8; ++g) {
      int j = g + 8 * hf;
      float raw = pacc[t][g] + p_b2[16 * t + 8 * hf + g];
      float Lv = (j < t) ? raw : ((j == t) ? __expf(raw) : 0.f);
      ca[g] = fmaf(anu[t], Lv, ca[g]);
      cm[g] = fmaf(mnu[t], Lv, cm[g]);
    }
  }
  float qa = 0.f, qm = 0.f;
#pragma unroll
  for (int g = 0; g < 8; ++g) {
    qa = fmaf(ca[g], ca[g], qa);
    qm = fmaf(cm[g], cm[g], qm);
  }
  qa += __shfl_xor(qa, 16, 32);
  qm += __shfl_xor(qm, 16, 32);

  float vv = v_acc[0] + v_b2[0];
  if (hf == 0) out[rb + l] = -0.5f * (qa - qm) + vv;
}

// ---- pre-pass: pack all weights into WMMA A-fragment order (f32 -> f16) ----
// One thread per (fragment, lane): writes 16 contiguous halves (32B).
__global__ void k_pack_frags(const float* __restrict__ nu_w1,
                             const float* __restrict__ v_w1,
                             const float* __restrict__ p_w1,
                             const float* __restrict__ nu_w2,
                             const float* __restrict__ v_w2,
                             const float* __restrict__ p_w2,
                             _Float16* __restrict__ wpack) {
  int idx = blockIdx.x * blockDim.x + threadIdx.x;
  if (idx >= 16 * FRAGS_PER_KB * 32) return;
  int f = idx >> 5, lane = idx & 31;
  int kb = f / FRAGS_PER_KB, r = f - kb * FRAGS_PER_KB;

  const float* src;
  int O, row, kbase;
  bool pad = false;
  if (r < 24) {                      // layer-1: m, tile-local, k4
    int m = r >> 3, rr = r & 7, tl = rr >> 2, k4 = rr & 3;
    src = (m == 0) ? nu_w1 : (m == 1) ? v_w1 : p_w1;
    O = 512; row = (2 * kb + tl) * 16 + (lane & 15); kbase = 32 * k4;
  } else if (r == 24) {              // nu head
    src = nu_w2; O = 16;  row = lane & 15;             kbase = 32 * kb;
  } else if (r == 25) {              // v head, zero-padded rows 1..15
    src = v_w2;  O = 1;   row = lane & 15;             kbase = 32 * kb;
    pad = (row != 0);
  } else {                           // P head tile t2
    int t2 = r - 26;
    src = p_w2;  O = 256; row = t2 * 16 + (lane & 15); kbase = 32 * kb;
  }

  _Float16* dst = wpack + (size_t)f * 512 + lane * 16;
  int khi = kbase + (lane >> 4) * 16;  // A-layout: halves 0..15 -> k..k+15 (contiguous)
#pragma unroll
  for (int i = 0; i < 16; ++i) {
    int k = khi + i;
    float v = pad ? 0.f : src[(size_t)k * O + row];
    dst[i] = (_Float16)v;
  }
}

extern "C" void kernel_launch(void* const* d_in, const int* in_sizes, int n_in,
                              void* d_out, int out_size, void* d_ws, size_t ws_size,
                              hipStream_t stream) {
  (void)in_sizes; (void)n_in; (void)out_size; (void)ws_size;
  const float* state  = (const float*)d_in[0];
  const float* action = (const float*)d_in[1];
  const float* nu_w1  = (const float*)d_in[2];
  const float* nu_b1  = (const float*)d_in[3];
  const float* nu_w2  = (const float*)d_in[4];
  const float* nu_b2  = (const float*)d_in[5];
  const float* v_w1   = (const float*)d_in[6];
  const float* v_b1   = (const float*)d_in[7];
  const float* v_w2   = (const float*)d_in[8];
  const float* v_b2   = (const float*)d_in[9];
  const float* p_w1   = (const float*)d_in[10];
  const float* p_b1   = (const float*)d_in[11];
  const float* p_w2   = (const float*)d_in[12];
  const float* p_b2   = (const float*)d_in[13];
  float* out = (float*)d_out;

  _Float16* wpack = (_Float16*)d_ws;   // 16 * 43008 B = 672 KB

  k_pack_frags<<<84, 256, 0, stream>>>(nu_w1, v_w1, p_w1, nu_w2, v_w2, p_w2, wpack);

  // 65536 rows / (8 waves * 16 rows) = 512 blocks; 84KB dynamic LDS
  qmodel_fused<<<512, 256, 2 * FRAGS_PER_KB * 1024, stream>>>(
      state, action, wpack,
      nu_b1, nu_b2, v_b1, v_b2, p_b1, p_b2, out);
}
